// DGCNNEncoder_56212531970648
// MI455X (gfx1250) — matrix operand order repaired and checked by reading, not compiled, not run on hardware
//
#include <hip/hip_runtime.h>
#include <hip/hip_bf16.h>

// VN-DGCNN encoder for MI455X (gfx1250, wave32, WMMA).
// All heavy GEMMs run as v_wmma_f32_16x16x32_f16 (f32 accumulate).
// EdgeConv uses the identity  Wp*[nbr-ctr;ctr] = Wp_lo*nbr + (Wp_hi-Wp_lo)*ctr
// so B operands are plain contiguous f16 feature columns (no branchy on-the-fly
// edge-feature build). Axis-pad columns of each quad carry garbage that the
// epilogues never read, so all B loads are unconditional b128s.

typedef _Float16 f16;
typedef __attribute__((ext_vector_type(16))) _Float16 v16h;
typedef __attribute__((ext_vector_type(8)))  float    v8f;

#define KNN 20
#define NB  8
#define NPTS 1024

__device__ __forceinline__ v8f wmma16(v16h a, v16h b, v8f c) {
  // 8 args: (neg_a, A, neg_b, B, c_mod, C, reuse_a, reuse_b)
  return __builtin_amdgcn_wmma_f32_16x16x32_f16(
      false, a, false, b, (short)0, c, false, false);
}

// A operand: 16x32 f16 (MxK). lanes 0-15: M=lane, K 0..15;
// lanes 16-31: same M, K 16..31. row already offset by lane&15.
__device__ __forceinline__ v16h load_a_row(const f16* __restrict__ row, int kk) {
  const int lane = threadIdx.x & 31;
  const int off  = (lane < 16) ? 0 : 16;
  v16h a;
#pragma unroll
  for (int t = 0; t < 16; ++t) a[t]     = row[kk + off + t];
  return a;
}

// B operand: 32x16 f16 (KxN). lane col = lane&15; halves = K koff..koff+15.
__device__ __forceinline__ v16h load_b_col(const f16* __restrict__ col, int kk) {
  const int lane = threadIdx.x & 31;
  const int koff = (lane < 16) ? 0 : 16;
  v16h b;
#pragma unroll
  for (int t = 0; t < 16; ++t) b[t] = col[kk + koff + t];
  return b;
}

// ---------------------------------------------------------------- staging ---
__global__ void fill_zero_f16_kernel(f16* __restrict__ p, size_t count) {
  size_t i = (size_t)blockIdx.x * blockDim.x + threadIdx.x;
  if (i < count) p[i] = (f16)0.f;
}

// dst[r][c] (OP x CP f16) = src[r*CinTotal + ofs + c] - (sub ? src[r*CinTotal + c] : 0)
// for r<O, c<Ccount; zero elsewhere.
__global__ void stage_weight_kernel(const float* __restrict__ src, f16* __restrict__ dst,
                                    int O, int CinTotal, int Ccount, int ofs, int sub,
                                    int OP, int CP) {
  int i = blockIdx.x * blockDim.x + threadIdx.x;
  if (i >= OP * CP) return;
  int c = i % CP, r = i / CP;
  float v = 0.f;
  if (r < O && c < Ccount) {
    v = src[r * CinTotal + ofs + c];
    if (sub) v -= src[r * CinTotal + c];
  }
  dst[i] = (f16)v;
}

// x:(B,N,3) f32 -> Xknn1[b][n][32] (d=c*3+axis, C=1) and Xct1[b][axis][n][32].
__global__ void stage_input_kernel(const float* __restrict__ x,
                                   f16* __restrict__ Xknn1, f16* __restrict__ Xct1) {
  int i = blockIdx.x * blockDim.x + threadIdx.x;
  if (i >= NB * NPTS) return;
  int n = i % NPTS, b = i / NPTS;
  float v[3];
  v[0] = x[(size_t)i * 3 + 0]; v[1] = x[(size_t)i * 3 + 1]; v[2] = x[(size_t)i * 3 + 2];
  f16* kp = Xknn1 + (size_t)i * 32;
#pragma unroll
  for (int d = 0; d < 32; ++d) kp[d] = (f16)0.f;
  kp[0] = (f16)v[0]; kp[1] = (f16)v[1]; kp[2] = (f16)v[2];
#pragma unroll
  for (int ax = 0; ax < 3; ++ax)
    Xct1[(((size_t)(b * 3 + ax)) * NPTS + n) * 32] = (f16)v[ax];   // pads prefilled 0
}

// ------------------------------------------------------------------- kNN ----
template<int DPAD>
__global__ void sumsq_kernel(const f16* __restrict__ X, float* __restrict__ sq, int total) {
  int i = blockIdx.x * blockDim.x + threadIdx.x;
  if (i >= total) return;
  const f16* p = X + (size_t)i * DPAD;
  float s = 0.f;
#pragma unroll 8
  for (int d = 0; d < DPAD; ++d) { float v = (float)p[d]; s += v * v; }
  sq[i] = s;
}

// Gram via WMMA: neg_dist[n][m] = 2*<f_n,f_m> - |f_n|^2 - |f_m|^2.
template<int DPAD>
__global__ void __launch_bounds__(32)
gram_knn_kernel(const f16* __restrict__ X, const float* __restrict__ sq,
                float* __restrict__ nd) {
  const int lane = threadIdx.x & 31;
  const int b  = blockIdx.z;
  const int i0 = blockIdx.x * 16;
  const int j0 = blockIdx.y * 16;
  const f16* Xb   = X + (size_t)b * NPTS * DPAD;
  const f16* arow = Xb + (size_t)(i0 + (lane & 15)) * DPAD;
  const f16* bcol = Xb + (size_t)(j0 + (lane & 15)) * DPAD;
  v8f c = {};
#pragma unroll
  for (int kk = 0; kk < DPAD; kk += 32)
    c = wmma16(load_a_row(arow, kk), load_b_col(bcol, kk), c);
  const int colN = j0 + (lane & 15);
  const int mhi  = (lane < 16) ? 0 : 8;
  const float sqc = sq[b * NPTS + colN];
#pragma unroll
  for (int r = 0; r < 8; ++r) {
    int row = i0 + r + mhi;
    nd[((size_t)b * NPTS + row) * NPTS + colN] = 2.f * c[r] - sq[b * NPTS + row] - sqc;
  }
}

__device__ __forceinline__ void topk_insert(float (&bv)[KNN], int (&bi)[KNN],
                                            float v, int m) {
  if (v > bv[KNN - 1]) {
    bv[KNN - 1] = v; bi[KNN - 1] = m;
#pragma unroll
    for (int t = KNN - 1; t > 0; --t) {
      if (bv[t] > bv[t - 1]) {
        float tv = bv[t]; bv[t] = bv[t - 1]; bv[t - 1] = tv;
        int   ti = bi[t]; bi[t] = bi[t - 1]; bi[t - 1] = ti;
      }
    }
  }
}

// Per-row top-20 by register insertion sort (self included, like jax.top_k).
__global__ void topk_kernel(const float* __restrict__ nd, int* __restrict__ knnIdx) {
  int i = blockIdx.x * blockDim.x + threadIdx.x;
  if (i >= NB * NPTS) return;
  const float4* row4 = (const float4*)(nd + (size_t)i * NPTS);
  float bv[KNN]; int bi[KNN];
#pragma unroll
  for (int t = 0; t < KNN; ++t) { bv[t] = -3.4e38f; bi[t] = 0; }
  for (int m4 = 0; m4 < NPTS / 4; ++m4) {
    float4 v4 = row4[m4];
    topk_insert(bv, bi, v4.x, 4 * m4 + 0);
    topk_insert(bv, bi, v4.y, 4 * m4 + 1);
    topk_insert(bv, bi, v4.z, 4 * m4 + 2);
    topk_insert(bv, bi, v4.w, 4 * m4 + 3);
  }
  int* o = knnIdx + (size_t)i * KNN;
#pragma unroll
  for (int t = 0; t < KNN; ++t) o[t] = bi[t];
}

// --------------------------------------------------------------- EdgeConv ---
// Columns: j = (n*K + k)*4 + axis, axis in {0,1,2,pad}. Pad columns compute
// garbage from the clamped axis plane; epilogue never reads them.
// P = WpN*X[:,nbr] + WpC*X[:,ctr]  with WpN=Wp_lo, WpC=Wp_hi-Wp_lo (same Wd).
template<int CPC, int OREAL, int OPB>
__global__ void __launch_bounds__(32)
edgeconv_kernel(const f16* __restrict__ Xct, const int* __restrict__ knnIdx,
                const f16* __restrict__ WpN, const f16* __restrict__ WpC,
                const f16* __restrict__ WdN, const f16* __restrict__ WdC,
                f16* __restrict__ Hct) {
  const int lane = threadIdx.x & 31;
  const int b  = blockIdx.z;
  const int o0 = blockIdx.y * 16;
  const int j0 = blockIdx.x * 16;
  const int NK = NPTS * KNN;

  const int j    = j0 + (lane & 15);
  const int quad = j >> 2;
  const int axis = j & 3;
  const int axc  = (axis < 3) ? axis : 2;       // clamp: pad column -> garbage, unread
  const int n    = quad / KNN;
  const int k    = quad - n * KNN;
  const int nbr  = knnIdx[((size_t)b * NPTS + n) * KNN + k];

  const f16* nbrCol = Xct + (((size_t)(b * 3 + axc)) * NPTS + nbr) * CPC;
  const f16* ctrCol = Xct + (((size_t)(b * 3 + axc)) * NPTS + n  ) * CPC;
  const int  m15    = lane & 15;
  const f16* apN = WpN + (size_t)(o0 + m15) * CPC;
  const f16* apC = WpC + (size_t)(o0 + m15) * CPC;
  const f16* adN = WdN + (size_t)(o0 + m15) * CPC;
  const f16* adC = WdC + (size_t)(o0 + m15) * CPC;

  v8f cP = {}, cD = {};
#pragma unroll
  for (int kk = 0; kk < CPC; kk += 32) {
    v16h bN = load_b_col(nbrCol, kk);
    v16h bC = load_b_col(ctrCol, kk);
    cP = wmma16(load_a_row(apN, kk), bN, cP);
    cP = wmma16(load_a_row(apC, kk), bC, cP);
    cD = wmma16(load_a_row(adN, kk), bN, cD);
    cD = wmma16(load_a_row(adC, kk), bC, cD);
  }

  __shared__ float lP[16][16];
  __shared__ float lD[16][16];
  const int mhi = (lane < 16) ? 0 : 8;
#pragma unroll
  for (int r = 0; r < 8; ++r) {
    lP[r + mhi][m15] = cP[r];
    lD[r + mhi][m15] = cD[r];
  }
  __syncthreads();

  // 16 rows x 4 quads = 64 VN-leaky items; 2 per lane. Only axes 0..2 read.
#pragma unroll
  for (int it = 0; it < 2; ++it) {
    int item = lane + it * 32;
    int m  = item >> 2;
    int ql = item & 3;
    int q2 = (j0 + ql * 4) >> 2;
    float p0 = lP[m][ql * 4 + 0], p1 = lP[m][ql * 4 + 1], p2 = lP[m][ql * 4 + 2];
    float d0 = lD[m][ql * 4 + 0], d1 = lD[m][ql * 4 + 1], d2 = lD[m][ql * 4 + 2];
    float dot = p0 * d0 + p1 * d1 + p2 * d2;
    float dsq = d0 * d0 + d1 * d1 + d2 * d2;
    float s   = dot / (dsq + 1e-6f);
    float h0 = p0, h1 = p1, h2 = p2;
    if (dot < 0.f) { h0 = p0 - s * d0; h1 = p1 - s * d1; h2 = p2 - s * d2; }
    h0 = 0.2f * p0 + 0.8f * h0;
    h1 = 0.2f * p1 + 0.8f * h1;
    h2 = 0.2f * p2 + 0.8f * h2;
    int  o    = o0 + m;
    bool live = (o < OREAL);
    Hct[(((size_t)(b * 3 + 0)) * NK + q2) * OPB + o] = (f16)(live ? h0 : 0.f);
    Hct[(((size_t)(b * 3 + 1)) * NK + q2) * OPB + o] = (f16)(live ? h1 : 0.f);
    Hct[(((size_t)(b * 3 + 2)) * NK + q2) * OPB + o] = (f16)(live ? h2 : 0.f);
  }
}

// Pool direction GEMM Dm = Wm x H, fused with dot_axis(H, Dm) -> dotbuf.
template<int OREAL, int OPB>
__global__ void __launch_bounds__(32)
pool_gemm_kernel(const f16* __restrict__ Hct, const f16* __restrict__ Wmf,
                 float* __restrict__ dotbuf) {
  const int lane = threadIdx.x & 31;
  const int b  = blockIdx.z;
  const int o0 = blockIdx.y * 16;
  const int j0 = blockIdx.x * 16;
  const int NK = NPTS * KNN;

  const int j    = j0 + (lane & 15);
  const int quad = j >> 2;
  const int axis = j & 3;
  const int axc  = (axis < 3) ? axis : 2;
  const f16* bcol = Hct + (((size_t)(b * 3 + axc)) * NK + quad) * OPB;
  const f16* aRow = Wmf + (size_t)(o0 + (lane & 15)) * OPB;

  v8f c = {};
#pragma unroll
  for (int kk = 0; kk < OPB; kk += 32)
    c = wmma16(load_a_row(aRow, kk), load_b_col(bcol, kk), c);

  __shared__ float lDm[16][16];
  const int mhi = (lane < 16) ? 0 : 8;
#pragma unroll
  for (int r = 0; r < 8; ++r) lDm[r + mhi][lane & 15] = c[r];
  __syncthreads();

#pragma unroll
  for (int it = 0; it < 2; ++it) {
    int item = lane + it * 32;
    int m  = item >> 2;
    int ql = item & 3;
    int q2 = (j0 + ql * 4) >> 2;
    int o  = o0 + m;
    if (o < OREAL) {
      float dot = 0.f;
#pragma unroll
      for (int ax = 0; ax < 3; ++ax) {
        float h = (float)Hct[(((size_t)(b * 3 + ax)) * NK + q2) * OPB + o];
        dot += h * lDm[m][ql * 4 + ax];
      }
      dotbuf[((size_t)(b * OREAL + o)) * NK + q2] = dot;
    }
  }
}

// Argmax over K and gather into next-layer staged layouts + concat buffer.
__global__ void pool_argmax_kernel(const float* __restrict__ dotbuf,
                                   const f16* __restrict__ Hct,
                                   int OREAL, int OPB,
                                   f16* __restrict__ XknnNext, int DPADn,
                                   f16* __restrict__ XctNext, int CSn,
                                   f16* __restrict__ Yct, int ycoff) {
  int i = blockIdx.x * blockDim.x + threadIdx.x;
  int total = NB * OREAL * NPTS;
  if (i >= total) return;
  int n  = i % NPTS;
  int bo = i / NPTS;
  int o  = bo % OREAL;
  int b  = bo / OREAL;
  const int NK = NPTS * KNN;
  const float* drow = dotbuf + ((size_t)(b * OREAL + o)) * NK + (size_t)n * KNN;
  float best = drow[0]; int bk = 0;
#pragma unroll
  for (int k = 1; k < KNN; ++k) { float v = drow[k]; if (v > best) { best = v; bk = k; } }
  int q = n * KNN + bk;
#pragma unroll
  for (int ax = 0; ax < 3; ++ax) {
    f16 h = Hct[(((size_t)(b * 3 + ax)) * NK + q) * OPB + o];
    if (XknnNext) XknnNext[((size_t)(b * NPTS) + n) * DPADn + o * 3 + ax] = h;
    if (XctNext)  XctNext[(((size_t)(b * 3 + ax)) * NPTS + n) * CSn + o]  = h;
    Yct[(((size_t)(b * 3 + ax)) * NPTS + n) * 192 + ycoff + o] = h;
  }
}

// ------------------------------------------------------------ final layer ---
__global__ void d5_kernel(const f16* __restrict__ Yct, const float* __restrict__ Wd5,
                          float* __restrict__ D5) {
  int i = blockIdx.x * blockDim.x + threadIdx.x;   // (b*3+axis)*N + n
  if (i >= NB * 3 * NPTS) return;
  const f16* col = Yct + (size_t)i * 192;
  float s = 0.f;
#pragma unroll 8
  for (int c = 0; c < 169; ++c) s += Wd5[c] * (float)col[c];
  D5[i] = s;
}

__global__ void __launch_bounds__(32)
gemm5_leaky_kernel(const f16* __restrict__ Yct, const f16* __restrict__ Wp5f,
                   const float* __restrict__ D5, float* __restrict__ Hfin) {
  const int lane = threadIdx.x & 31;
  const int b  = blockIdx.z;
  const int o0 = blockIdx.y * 16;
  const int j0 = blockIdx.x * 16;

  const int j    = j0 + (lane & 15);
  const int n    = j >> 2;
  const int axis = j & 3;
  const int axc  = (axis < 3) ? axis : 2;
  const f16* bcol = Yct + (((size_t)(b * 3 + axc)) * NPTS + n) * 192;
  const f16* aRow = Wp5f + (size_t)(o0 + (lane & 15)) * 192;

  v8f c = {};
#pragma unroll
  for (int kk = 0; kk < 192; kk += 32)
    c = wmma16(load_a_row(aRow, kk), load_b_col(bcol, kk), c);

  __shared__ float lP[16][16];
  const int mhi = (lane < 16) ? 0 : 8;
#pragma unroll
  for (int r = 0; r < 8; ++r) lP[r + mhi][lane & 15] = c[r];
  __syncthreads();

#pragma unroll
  for (int it = 0; it < 2; ++it) {
    int item = lane + it * 32;
    int m  = item >> 2;
    int ql = item & 3;
    int nn = (j0 + ql * 4) >> 2;
    int o  = o0 + m;
    if (o < 341) {
      float p0 = lP[m][ql * 4 + 0], p1 = lP[m][ql * 4 + 1], p2 = lP[m][ql * 4 + 2];
      float d0 = D5[(b * 3 + 0) * NPTS + nn];
      float d1 = D5[(b * 3 + 1) * NPTS + nn];
      float d2 = D5[(b * 3 + 2) * NPTS + nn];
      float dot = p0 * d0 + p1 * d1 + p2 * d2;
      float dsq = d0 * d0 + d1 * d1 + d2 * d2;
      float s   = dot / (dsq + 1e-6f);
      float h0 = p0, h1 = p1, h2 = p2;
      if (dot < 0.f) { h0 = p0 - s * d0; h1 = p1 - s * d1; h2 = p2 - s * d2; }
      h0 = 0.2f * p0 + 0.8f * h0;
      h1 = 0.2f * p1 + 0.8f * h1;
      h2 = 0.2f * p2 + 0.8f * h2;
      Hfin[((size_t)b * 1023 + (o * 3 + 0)) * NPTS + nn] = h0;
      Hfin[((size_t)b * 1023 + (o * 3 + 1)) * NPTS + nn] = h1;
      Hfin[((size_t)b * 1023 + (o * 3 + 2)) * NPTS + nn] = h2;
    }
  }
}

// Wave-per-row max + mean over N.
__global__ void __launch_bounds__(32)
reduce_kernel(const float* __restrict__ Hfin, float* __restrict__ g) {
  const int b    = blockIdx.y;
  const int oc   = blockIdx.x;          // 0..1022
  const int lane = threadIdx.x;
  const float* row = Hfin + ((size_t)b * 1023 + oc) * NPTS;
  float mx = -3.4e38f, sm = 0.f;
  for (int n = lane; n < NPTS; n += 32) { float v = row[n]; mx = fmaxf(mx, v); sm += v; }
#pragma unroll
  for (int s = 16; s > 0; s >>= 1) {
    mx = fmaxf(mx, __shfl_xor(mx, s, 32));
    sm += __shfl_xor(sm, s, 32);
  }
  if (lane == 0) {
    g[b * 2046 + oc]        = mx;
    g[b * 2046 + 1023 + oc] = sm * (1.f / (float)NPTS);
  }
}

__global__ void fc_kernel(const float* __restrict__ in, const float* __restrict__ W,
                          const float* __restrict__ bias, float* __restrict__ out,
                          int IN, int OUT, float slope, int applyLeaky) {
  int i = blockIdx.x * blockDim.x + threadIdx.x;
  if (i >= NB * OUT) return;
  int o = i % OUT, b = i / OUT;
  const float* ip = in + (size_t)b * IN;
  const float* wp = W + (size_t)o * IN;
  float s = bias[o];
#pragma unroll 4
  for (int c = 0; c < IN; ++c) s += ip[c] * wp[c];
  if (applyLeaky && s < 0.f) s *= slope;
  out[i] = s;
}

// ------------------------------------------------------------------ host ----
static inline size_t alignUp256(size_t x) { return (x + 255) & ~(size_t)255; }

extern "C" void kernel_launch(void* const* d_in, const int* in_sizes, int n_in,
                              void* d_out, int out_size, void* d_ws, size_t ws_size,
                              hipStream_t stream) {
  (void)in_sizes; (void)n_in; (void)out_size; (void)ws_size;
  const float* x   = (const float*)d_in[0];
  const float* Wp1 = (const float*)d_in[1];
  const float* Wd1 = (const float*)d_in[2];
  const float* Wm1 = (const float*)d_in[3];
  const float* Wp2 = (const float*)d_in[4];
  const float* Wd2 = (const float*)d_in[5];
  const float* Wm2 = (const float*)d_in[6];
  const float* Wp3 = (const float*)d_in[7];
  const float* Wd3 = (const float*)d_in[8];
  const float* Wm3 = (const float*)d_in[9];
  const float* Wp4 = (const float*)d_in[10];
  const float* Wd4 = (const float*)d_in[11];
  const float* Wm4 = (const float*)d_in[12];
  const float* Wp5 = (const float*)d_in[13];
  const float* Wd5 = (const float*)d_in[14];
  const float* W1  = (const float*)d_in[15];
  const float* b1  = (const float*)d_in[16];
  const float* W2  = (const float*)d_in[17];
  const float* b2  = (const float*)d_in[18];
  const float* W3  = (const float*)d_in[19];
  const float* b3  = (const float*)d_in[20];

  const int B = NB, N = NPTS;
  const size_t NK = (size_t)N * KNN;

  char*  ws  = (char*)d_ws;
  size_t off = 0;
  auto alloc = [&](size_t bytes) { void* p = ws + off; off = alignUp256(off + bytes); return p; };

  float* ndist  = (float*)alloc((size_t)B * N * N * 4);
  int*   knnIdx = (int*)  alloc((size_t)B * N * KNN * 4);
  float* sqb    = (float*)alloc((size_t)B * N * 4);
  f16*   Xknn1  = (f16*)  alloc((size_t)B * N * 32 * 2);
  f16*   Xknn2  = (f16*)  alloc((size_t)B * N * 64 * 2);
  f16*   Xknn3  = (f16*)  alloc((size_t)B * N * 64 * 2);
  f16*   Xknn4  = (f16*)  alloc((size_t)B * N * 128 * 2);
  f16*   Xct1   = (f16*)  alloc((size_t)B * 3 * N * 32 * 2);
  f16*   Xct2   = (f16*)  alloc((size_t)B * 3 * N * 32 * 2);
  f16*   Xct3   = (f16*)  alloc((size_t)B * 3 * N * 32 * 2);
  f16*   Xct4   = (f16*)  alloc((size_t)B * 3 * N * 64 * 2);
  f16*   Yct    = (f16*)  alloc((size_t)B * 3 * N * 192 * 2);
  f16*   Hct    = (f16*)  alloc((size_t)B * 3 * NK * 96 * 2);   // reused, max OPB=96
  float* dotbuf = (float*)alloc((size_t)B * 96 * NK * 4);       // reused, max O=85
  // split edge-conv weights: N = lo part, C = hi - lo
  f16* Wp1Nf = (f16*)alloc(32 * 32 * 2);  f16* Wp1Cf = (f16*)alloc(32 * 32 * 2);
  f16* Wd1Nf = (f16*)alloc(32 * 32 * 2);  f16* Wd1Cf = (f16*)alloc(32 * 32 * 2);
  f16* Wm1f  = (f16*)alloc(32 * 32 * 2);
  f16* Wp2Nf = (f16*)alloc(32 * 32 * 2);  f16* Wp2Cf = (f16*)alloc(32 * 32 * 2);
  f16* Wd2Nf = (f16*)alloc(32 * 32 * 2);  f16* Wd2Cf = (f16*)alloc(32 * 32 * 2);
  f16* Wm2f  = (f16*)alloc(32 * 32 * 2);
  f16* Wp3Nf = (f16*)alloc(64 * 32 * 2);  f16* Wp3Cf = (f16*)alloc(64 * 32 * 2);
  f16* Wd3Nf = (f16*)alloc(64 * 32 * 2);  f16* Wd3Cf = (f16*)alloc(64 * 32 * 2);
  f16* Wm3f  = (f16*)alloc(64 * 64 * 2);
  f16* Wp4Nf = (f16*)alloc(96 * 64 * 2);  f16* Wp4Cf = (f16*)alloc(96 * 64 * 2);
  f16* Wd4Nf = (f16*)alloc(96 * 64 * 2);  f16* Wd4Cf = (f16*)alloc(96 * 64 * 2);
  f16* Wm4f  = (f16*)alloc(96 * 96 * 2);
  f16* Wp5f  = (f16*)alloc(352 * 192 * 2);
  float* D5     = (float*)alloc((size_t)B * 3 * N * 4);
  float* Hfin   = (float*)alloc((size_t)B * 1023 * N * 4);
  float* gbuf   = (float*)alloc((size_t)B * 2046 * 4);
  float* f1buf  = (float*)alloc((size_t)B * 512 * 4);
  float* f2buf  = (float*)alloc((size_t)B * 256 * 4);

  auto fillz = [&](f16* p, size_t n) {
    fill_zero_f16_kernel<<<dim3((unsigned)((n + 255) / 256)), 256, 0, stream>>>(p, n);
  };
  // zero pads of staged feature buffers (real entries rewritten every call)
  fillz(Xknn2, (size_t)B * N * 64);
  fillz(Xknn3, (size_t)B * N * 64);
  fillz(Xknn4, (size_t)B * N * 128);
  fillz(Xct1,  (size_t)B * 3 * N * 32);
  fillz(Xct2,  (size_t)B * 3 * N * 32);
  fillz(Xct3,  (size_t)B * 3 * N * 32);
  fillz(Xct4,  (size_t)B * 3 * N * 64);
  fillz(Yct,   (size_t)B * 3 * N * 192);

  auto stageW = [&](const float* src, f16* dst, int O, int CinTotal, int Ccount,
                    int ofs, int sub, int OP, int CP) {
    int n = OP * CP;
    stage_weight_kernel<<<(n + 255) / 256, 256, 0, stream>>>(src, dst, O, CinTotal,
                                                             Ccount, ofs, sub, OP, CP);
  };
  // layer 1: C=1 (2C=2)
  stageW(Wp1, Wp1Nf, 21, 2, 1, 0, 0, 32, 32);  stageW(Wp1, Wp1Cf, 21, 2, 1, 1, 1, 32, 32);
  stageW(Wd1, Wd1Nf, 21, 2, 1, 0, 0, 32, 32);  stageW(Wd1, Wd1Cf, 21, 2, 1, 1, 1, 32, 32);
  stageW(Wm1, Wm1f,  21, 21, 21, 0, 0, 32, 32);
  // layer 2: C=21
  stageW(Wp2, Wp2Nf, 21, 42, 21, 0, 0, 32, 32); stageW(Wp2, Wp2Cf, 21, 42, 21, 21, 1, 32, 32);
  stageW(Wd2, Wd2Nf, 21, 42, 21, 0, 0, 32, 32); stageW(Wd2, Wd2Cf, 21, 42, 21, 21, 1, 32, 32);
  stageW(Wm2, Wm2f,  21, 21, 21, 0, 0, 32, 32);
  // layer 3: C=21, O=42
  stageW(Wp3, Wp3Nf, 42, 42, 21, 0, 0, 64, 32); stageW(Wp3, Wp3Cf, 42, 42, 21, 21, 1, 64, 32);
  stageW(Wd3, Wd3Nf, 42, 42, 21, 0, 0, 64, 32); stageW(Wd3, Wd3Cf, 42, 42, 21, 21, 1, 64, 32);
  stageW(Wm3, Wm3f,  42, 42, 42, 0, 0, 64, 64);
  // layer 4: C=42, O=85
  stageW(Wp4, Wp4Nf, 85, 84, 42, 0, 0, 96, 64); stageW(Wp4, Wp4Cf, 85, 84, 42, 42, 1, 96, 64);
  stageW(Wd4, Wd4Nf, 85, 84, 42, 0, 0, 96, 64); stageW(Wd4, Wd4Cf, 85, 84, 42, 42, 1, 96, 64);
  stageW(Wm4, Wm4f,  85, 85, 85, 0, 0, 96, 96);
  // final VN layer
  stageW(Wp5, Wp5f, 341, 169, 169, 0, 0, 352, 192);

  stage_input_kernel<<<(B * N + 255) / 256, 256, 0, stream>>>(x, Xknn1, Xct1);

  const dim3 gramGrid(N / 16, N / 16, B);
  const unsigned colTiles = (unsigned)(4 * NK / 16);   // 5120
  const int bn = B * N;

  // ---------------- layer 1: C=1 -> O=21 ----------------
  sumsq_kernel<32><<<(bn + 255) / 256, 256, 0, stream>>>(Xknn1, sqb, bn);
  gram_knn_kernel<32><<<gramGrid, 32, 0, stream>>>(Xknn1, sqb, ndist);
  topk_kernel<<<(bn + 127) / 128, 128, 0, stream>>>(ndist, knnIdx);
  edgeconv_kernel<32, 21, 32><<<dim3(colTiles, 2, B), 32, 0, stream>>>(
      Xct1, knnIdx, Wp1Nf, Wp1Cf, Wd1Nf, Wd1Cf, Hct);
  pool_gemm_kernel<21, 32><<<dim3(colTiles, 2, B), 32, 0, stream>>>(Hct, Wm1f, dotbuf);
  pool_argmax_kernel<<<(B * 21 * N + 255) / 256, 256, 0, stream>>>(
      dotbuf, Hct, 21, 32, Xknn2, 64, Xct2, 32, Yct, 0);

  // ---------------- layer 2: C=21 -> O=21 ----------------
  sumsq_kernel<64><<<(bn + 255) / 256, 256, 0, stream>>>(Xknn2, sqb, bn);
  gram_knn_kernel<64><<<gramGrid, 32, 0, stream>>>(Xknn2, sqb, ndist);
  topk_kernel<<<(bn + 127) / 128, 128, 0, stream>>>(ndist, knnIdx);
  edgeconv_kernel<32, 21, 32><<<dim3(colTiles, 2, B), 32, 0, stream>>>(
      Xct2, knnIdx, Wp2Nf, Wp2Cf, Wd2Nf, Wd2Cf, Hct);
  pool_gemm_kernel<21, 32><<<dim3(colTiles, 2, B), 32, 0, stream>>>(Hct, Wm2f, dotbuf);
  pool_argmax_kernel<<<(B * 21 * N + 255) / 256, 256, 0, stream>>>(
      dotbuf, Hct, 21, 32, Xknn3, 64, Xct3, 32, Yct, 21);

  // ---------------- layer 3: C=21 -> O=42 ----------------
  sumsq_kernel<64><<<(bn + 255) / 256, 256, 0, stream>>>(Xknn3, sqb, bn);
  gram_knn_kernel<64><<<gramGrid, 32, 0, stream>>>(Xknn3, sqb, ndist);
  topk_kernel<<<(bn + 127) / 128, 128, 0, stream>>>(ndist, knnIdx);
  edgeconv_kernel<32, 42, 64><<<dim3(colTiles, 4, B), 32, 0, stream>>>(
      Xct3, knnIdx, Wp3Nf, Wp3Cf, Wd3Nf, Wd3Cf, Hct);
  pool_gemm_kernel<42, 64><<<dim3(colTiles, 4, B), 32, 0, stream>>>(Hct, Wm3f, dotbuf);
  pool_argmax_kernel<<<(B * 42 * N + 255) / 256, 256, 0, stream>>>(
      dotbuf, Hct, 42, 64, Xknn4, 128, Xct4, 64, Yct, 42);

  // ---------------- layer 4: C=42 -> O=85 ----------------
  sumsq_kernel<128><<<(bn + 255) / 256, 256, 0, stream>>>(Xknn4, sqb, bn);
  gram_knn_kernel<128><<<gramGrid, 32, 0, stream>>>(Xknn4, sqb, ndist);
  topk_kernel<<<(bn + 127) / 128, 128, 0, stream>>>(ndist, knnIdx);
  edgeconv_kernel<64, 85, 96><<<dim3(colTiles, 6, B), 32, 0, stream>>>(
      Xct4, knnIdx, Wp4Nf, Wp4Cf, Wd4Nf, Wd4Cf, Hct);
  pool_gemm_kernel<85, 96><<<dim3(colTiles, 6, B), 32, 0, stream>>>(Hct, Wm4f, dotbuf);
  pool_argmax_kernel<<<(B * 85 * N + 255) / 256, 256, 0, stream>>>(
      dotbuf, Hct, 85, 96, (f16*)nullptr, 0, (f16*)nullptr, 0, Yct, 84);

  // ---------------- layer 5 + global pooling ----------------
  d5_kernel<<<(B * 3 * N + 255) / 256, 256, 0, stream>>>(Yct, Wd5, D5);
  gemm5_leaky_kernel<<<dim3(4 * N / 16, 352 / 16, B), 32, 0, stream>>>(Yct, Wp5f, D5, Hfin);
  reduce_kernel<<<dim3(1023, B), 32, 0, stream>>>(Hfin, gbuf);

  // ---------------- MLP head ----------------
  fc_kernel<<<(B * 512 + 255) / 256, 256, 0, stream>>>(gbuf,  W1, b1, f1buf, 2046, 512, 0.01f, 1);
  fc_kernel<<<(B * 256 + 255) / 256, 256, 0, stream>>>(f1buf, W2, b2, f2buf, 512,  256, 0.01f, 1);
  fc_kernel<<<(B * 128 + 255) / 256, 256, 0, stream>>>(f2buf, W3, b3, (float*)d_out, 256, 128, 0.01f, 0);
}
